// GGATUNet_74766790689285
// MI455X (gfx1250) — matrix operand order repaired and checked by reading, not compiled, not run on hardware
//
#include <hip/hip_runtime.h>
#include <hip/hip_bf16.h>

// ---------------------------------------------------------------------------
// CDNA5 (gfx1250) GGAT-U-Net forward pass.
// Dense GEMMs: v_wmma_f32_16x16x32_bf16 (wave32), 256-thread blocks (8 waves),
// block tile 128x64. B operand staged global -> LDS (wide loads, K-pair
// interleaved) then read as bank-conflict-free ds_load_b32 fragments.
// f32 accumulation, bf16 operands.
// ---------------------------------------------------------------------------

typedef __attribute__((ext_vector_type(16))) __bf16 v16bf;
typedef __attribute__((ext_vector_type(8)))  float  v8f;

union V16U { v16bf v; unsigned int u[8]; };

__device__ __forceinline__ unsigned short f2bf(float f) {
  unsigned int u = __float_as_uint(f);
  u += 0x7FFFu + ((u >> 16) & 1u);      // round-to-nearest-even
  return (unsigned short)(u >> 16);
}

// ------------------------------ elementwise --------------------------------
__global__ void k_zero_f32(float* p, long n) {
  long i = (long)blockIdx.x * blockDim.x + threadIdx.x;
  if (i < n) p[i] = 0.f;
}

__global__ void k_scatter_edges(float* A, const int* ei, int E, int n) {
  int e = blockIdx.x * blockDim.x + threadIdx.x;
  if (e < E) {
    int r = ei[e];
    int c = ei[E + e];
    A[(long)r * n + c] = 1.0f;
  }
}

__global__ void k_to_bf16(const float* x, unsigned short* y, long n) {
  long i = (long)blockIdx.x * blockDim.x + threadIdx.x;
  if (i < n) y[i] = f2bf(x[i]);
}

__global__ void k_addI_bf16(const float* A, unsigned short* Ah, int n) {
  long i = (long)blockIdx.x * blockDim.x + threadIdx.x;
  long t = (long)n * n;
  if (i < t) {
    int r = (int)(i / n), c = (int)(i % n);
    float v = A[i] + (r == c ? 1.f : 0.f);
    Ah[i] = f2bf(v);
  }
}

// one block (256 thr) per row: deg = sum(row) (+binarize) + 1 ; dinv = rsqrt
__global__ void k_deg_rsqrt(const float* A, float* dinv, int n, int binarize) {
  __shared__ float red[256];
  int i = blockIdx.x;
  float s = 0.f;
  for (int j = threadIdx.x; j < n; j += 256) {
    float v = A[(long)i * n + j];
    s += binarize ? (v > 0.f ? 1.f : 0.f) : v;
  }
  red[threadIdx.x] = s;
  __syncthreads();
  for (int w = 128; w > 0; w >>= 1) {
    if (threadIdx.x < w) red[threadIdx.x] += red[threadIdx.x + w];
    __syncthreads();
  }
  if (threadIdx.x == 0) {
    float d = red[0] + 1.0f;                 // self loop
    dinv[i] = d > 0.f ? rsqrtf(d) : 0.f;
  }
}

// An = dinv_r * (binarize?(A>0):A + I) * dinv_c   -> bf16
__global__ void k_norm_bf16(const float* A, const float* dinv,
                            unsigned short* An, int n, int binarize) {
  long i = (long)blockIdx.x * blockDim.x + threadIdx.x;
  long t = (long)n * n;
  if (i < t) {
    int r = (int)(i / n), c = (int)(i % n);
    float v = A[i];
    if (binarize) v = v > 0.f ? 1.f : 0.f;
    if (r == c) v += 1.f;
    An[i] = f2bf(dinv[r] * v * dinv[c]);
  }
}

__global__ void k_elu(float* x, long n) {
  long i = (long)blockIdx.x * blockDim.x + threadIdx.x;
  if (i < n) {
    float v = x[i];
    x[i] = v > 0.f ? v : (expf(v) - 1.f);
  }
}

__global__ void k_add(float* a, const float* b, long n) {
  long i = (long)blockIdx.x * blockDim.x + threadIdx.x;
  if (i < n) a[i] += b[i];
}

// W[H,F,O] f32 -> Wc[F, H*O] bf16 (head-concat along columns)
__global__ void k_pack_w(const float* W, unsigned short* Wc, int H, int F, int O) {
  long idx = (long)blockIdx.x * blockDim.x + threadIdx.x;
  long total = (long)H * F * O;
  if (idx < total) {
    int h = (int)(idx / ((long)F * O));
    int rem = (int)(idx % ((long)F * O));
    int f = rem / O, o = rem % O;
    Wc[(long)f * (H * O) + h * O + o] = f2bf(W[idx]);
  }
}

// --------------------------- WMMA bf16 GEMM --------------------------------
// C[M,N] f32 = A[M,K] bf16 * B[K,N] bf16 (+bias[N]) (optionally zero diag)
// Block = 256 threads (8 waves). Block tile 128x64; each wave owns a 16x64
// strip (4 accumulators, A fragment shared across the 4 column tiles).
// B tile (32x64 bf16) staged in LDS, K-pair interleaved:
//   ldsB[t*64 + c] = B[kk+2t, c] | B[kk+2t+1, c] << 16
// so a lane's fragment word j is ds_load_b32 at [ ((j>>2)*8+half*4+(j&3))*64 + col ].
// Requires: M % 128 == 0, N % 64 == 0, K % 32 == 0 (true for all calls here).
__global__ __launch_bounds__(256)
void k_gemm_bf16(const unsigned short* __restrict__ A,
                 const unsigned short* __restrict__ B,
                 float* __restrict__ C,
                 const float* __restrict__ bias,
                 int M, int N, int K, int zero_diag) {
  __shared__ unsigned int ldsB[16 * 64];   // 4 KB

  int tid  = threadIdx.x;
  int wave = tid >> 5;
  int lane = tid & 31;
  int half = lane >> 4;
  int l15  = lane & 15;

  int rowBase = blockIdx.x * 128 + wave * 16;
  int colBase = blockIdx.y * 64;

  const unsigned short* Arow = A + (long)(rowBase + l15) * K;
  int col0 = colBase + l15;

  // cooperative B-load coords: lt = K-pair row (0..15), lg = 4-col group (0..15)
  int lt = tid >> 4;
  int lg = tid & 15;

  v8f acc0 = {}, acc1 = {}, acc2 = {}, acc3 = {};
  for (int kk = 0; kk < K; kk += 32) {
    __syncthreads();                       // previous iteration's reads done
    {
      const unsigned short* b0p = B + (long)(kk + 2 * lt) * N + colBase + 4 * lg;
      const unsigned short* b1p = b0p + N;
      ushort4 r0 = *(const ushort4*)b0p;   // 8B wide load
      ushort4 r1 = *(const ushort4*)b1p;
      unsigned int* dst = &ldsB[lt * 64 + 4 * lg];
      dst[0] = (unsigned int)r0.x | ((unsigned int)r1.x << 16);
      dst[1] = (unsigned int)r0.y | ((unsigned int)r1.y << 16);
      dst[2] = (unsigned int)r0.z | ((unsigned int)r1.z << 16);
      dst[3] = (unsigned int)r0.w | ((unsigned int)r1.w << 16);
    }
    __syncthreads();

    V16U a;
#pragma unroll
    for (int j = 0; j < 8; ++j) {
      // A 16x32 bf16 fragment: VGPR j holds K = ka, ka+1 for row (lane&15)
      int ka = ((j >> 2) << 4) + (half << 3) + ((j & 3) << 1);
      a.u[j] = *(const unsigned int*)(Arow + kk + ka);
    }
    V16U b0, b1, b2, b3;
#pragma unroll
    for (int j = 0; j < 8; ++j) {
      // B 32x16 fragment word j: K-pair t = (j>>2)*8 + half*4 + (j&3), column l15 (+16c)
      const unsigned int* row = &ldsB[(((j >> 2) << 3) + (half << 2) + (j & 3)) * 64 + l15];
      b0.u[j] = row[0];
      b1.u[j] = row[16];
      b2.u[j] = row[32];
      b3.u[j] = row[48];
    }
    acc0 = __builtin_amdgcn_wmma_f32_16x16x32_bf16(false, a.v, false, b0.v,
                                                   (short)0, acc0, false, false);
    acc1 = __builtin_amdgcn_wmma_f32_16x16x32_bf16(false, a.v, false, b1.v,
                                                   (short)0, acc1, false, false);
    acc2 = __builtin_amdgcn_wmma_f32_16x16x32_bf16(false, a.v, false, b2.v,
                                                   (short)0, acc2, false, false);
    acc3 = __builtin_amdgcn_wmma_f32_16x16x32_bf16(false, a.v, false, b3.v,
                                                   (short)0, acc3, false, false);
  }

  // epilogue: bias hoisted (one load per column tile), branch-free diag mask
#pragma unroll
  for (int c = 0; c < 4; ++c) {
    v8f acc = (c == 0) ? acc0 : (c == 1) ? acc1 : (c == 2) ? acc2 : acc3;
    int col = col0 + c * 16;
    float bv = bias ? bias[col] : 0.f;
    float* Ccol = C + col;
    int row0 = rowBase + (half << 3);
#pragma unroll
    for (int v = 0; v < 8; ++v) {
      int row = row0 + v;
      float val = acc[v] + bv;
      val = (zero_diag & (row == col)) ? 0.f : val;
      Ccol[(long)row * N] = val;
    }
  }
}

// ------------------------------ gating -------------------------------------
// concat: z[n, H*O] *= sigmoid(<z[i,h,:], att[h,:]>), one block per (i,h), O=128
__global__ void k_gate_concat(float* z, const float* att, int n, int H, int O) {
  __shared__ float red[128];
  int b = blockIdx.x;
  int i = b / H, h = b % H;
  int o = threadIdx.x;
  long base = (long)i * H * O + (long)h * O;
  float v = (o < O) ? z[base + o] * att[h * O + o] : 0.f;
  red[threadIdx.x] = v;
  __syncthreads();
  for (int w = 64; w > 0; w >>= 1) {
    if (threadIdx.x < w) red[threadIdx.x] += red[threadIdx.x + w];
    __syncthreads();
  }
  float g = 1.f / (1.f + expf(-red[0]));
  if (o < O) z[base + o] *= g;
}

// mean over heads (final layer, O=16): out[i,o] = mean_h z[i,h,o]*gate(i,h)
__global__ void k_gate_mean(const float* z, const float* att, float* out,
                            int n, int H, int O) {
  int i = blockIdx.x * blockDim.x + threadIdx.x;
  if (i >= n) return;
  float acc[16];
  for (int o = 0; o < O; ++o) acc[o] = 0.f;
  for (int h = 0; h < H; ++h) {
    float dot = 0.f;
    for (int o = 0; o < O; ++o) dot += z[(long)i * H * O + h * O + o] * att[h * O + o];
    float g = 1.f / (1.f + expf(-dot));
    for (int o = 0; o < O; ++o) acc[o] += z[(long)i * H * O + h * O + o] * g;
  }
  for (int o = 0; o < O; ++o) out[(long)i * O + o] = acc[o] / (float)H;
}

// ---------------------------- top-k pooling --------------------------------
__global__ void k_pnorm(const float* p, float* outn, int Dd) {
  __shared__ float red[256];
  float s = 0.f;
  for (int i = threadIdx.x; i < Dd; i += 256) { float v = p[i]; s += v * v; }
  red[threadIdx.x] = s;
  __syncthreads();
  for (int w = 128; w > 0; w >>= 1) {
    if (threadIdx.x < w) red[threadIdx.x] += red[threadIdx.x + w];
    __syncthreads();
  }
  if (threadIdx.x == 0) outn[0] = sqrtf(red[0]);
}

__global__ void k_score(const float* x, const float* p, const float* pn,
                        float* s, int n, int Dd) {
  __shared__ float red[256];
  int i = blockIdx.x;
  float t = 0.f;
  for (int j = threadIdx.x; j < Dd; j += 256) t += x[(long)i * Dd + j] * p[j];
  red[threadIdx.x] = t;
  __syncthreads();
  for (int w = 128; w > 0; w >>= 1) {
    if (threadIdx.x < w) red[threadIdx.x] += red[threadIdx.x + w];
    __syncthreads();
  }
  if (threadIdx.x == 0) s[i] = red[0] / pn[0];
}

#define TOPK_NP 4096
// single-block LDS bitonic sort, descending score, index tie-break ascending
__global__ void k_topk(const float* score, int n, int k, int* perm, float* sval) {
  __shared__ float s[TOPK_NP];
  __shared__ int   id[TOPK_NP];
  for (int i = threadIdx.x; i < TOPK_NP; i += blockDim.x) {
    if (i < n) { s[i] = score[i]; id[i] = i; }
    else       { s[i] = -3.0e38f; id[i] = TOPK_NP + i; }
  }
  __syncthreads();
  for (int kk = 2; kk <= TOPK_NP; kk <<= 1) {
    for (int j = kk >> 1; j > 0; j >>= 1) {
      for (int i = threadIdx.x; i < TOPK_NP; i += blockDim.x) {
        int ixj = i ^ j;
        if (ixj > i) {
          bool descBlock = ((i & kk) == 0);
          float si = s[i], sj = s[ixj];
          int   ii = id[i], ij = id[ixj];
          bool iBefore = (si > sj) || (si == sj && ii < ij);
          bool doSwap = descBlock ? !iBefore : iBefore;
          if (doSwap) { s[i] = sj; s[ixj] = si; id[i] = ij; id[ixj] = ii; }
        }
      }
      __syncthreads();
    }
  }
  for (int i = threadIdx.x; i < k; i += blockDim.x) {
    perm[i] = id[i];
    sval[i] = s[i];
  }
}

__global__ void k_gather_x(const float* x, const int* perm, const float* sval,
                           float* xp, int k, int Dd) {
  int j = blockIdx.x;
  int r = perm[j];
  float t = tanhf(sval[j]);
  for (int c = threadIdx.x; c < Dd; c += blockDim.x)
    xp[(long)j * Dd + c] = x[(long)r * Dd + c] * t;
}

__global__ void k_gather_A(const float* A, const int* perm, float* Ap,
                           int k, int n) {
  int j1 = blockIdx.x;
  int r = perm[j1];
  for (int j2 = threadIdx.x; j2 < k; j2 += blockDim.x)
    Ap[(long)j1 * k + j2] = A[(long)r * n + perm[j2]];
}

__global__ void k_scatter_rows(float* up, const int* perm, const float* xs,
                               int k, int Dd) {
  int j = blockIdx.x;
  int r = perm[j];
  for (int c = threadIdx.x; c < Dd; c += blockDim.x)
    up[(long)r * Dd + c] = xs[(long)j * Dd + c];
}

// ---------------------------------------------------------------------------
extern "C" void kernel_launch(void* const* d_in, const int* in_sizes, int n_in,
                              void* d_out, int out_size, void* d_ws, size_t ws_size,
                              hipStream_t stream) {
  (void)in_sizes; (void)n_in; (void)out_size;

  const int N = 3072, E = N * 32, F_IN = 256, HID = 128, HEADS = 4, D = 512;
  const int P1 = N / 2, P2 = N / 4, P3 = N / 8;   // 1536, 768, 384

  const float* x   = (const float*)d_in[0];
  const int*   ei  = (const int*)  d_in[1];
  const float* Wc  = (const float*)d_in[2];
  const float* bc  = (const float*)d_in[3];
  const float* ac  = (const float*)d_in[4];
  const float* Wd1 = (const float*)d_in[5];
  const float* bd1 = (const float*)d_in[6];
  const float* ad1 = (const float*)d_in[7];
  const float* pd1 = (const float*)d_in[8];
  const float* Wd2 = (const float*)d_in[9];
  const float* bd2 = (const float*)d_in[10];
  const float* ad2 = (const float*)d_in[11];
  const float* pd2 = (const float*)d_in[12];
  const float* Wd3 = (const float*)d_in[13];
  const float* bd3 = (const float*)d_in[14];
  const float* ad3 = (const float*)d_in[15];
  const float* pd3 = (const float*)d_in[16];
  const float* Wu1 = (const float*)d_in[17];
  const float* bu1 = (const float*)d_in[18];
  const float* au1 = (const float*)d_in[19];
  const float* Wu2 = (const float*)d_in[20];
  const float* bu2 = (const float*)d_in[21];
  const float* au2 = (const float*)d_in[22];
  const float* Wu3 = (const float*)d_in[23];
  const float* bu3 = (const float*)d_in[24];
  const float* au3 = (const float*)d_in[25];

  // ---- carve workspace ----
  char* base = (char*)d_ws;
  size_t off = 0;
  auto carve = [&](size_t bytes) -> void* {
    void* p = base + off;
    off += (bytes + 255) & ~(size_t)255;
    return p;
  };
  float*          A1f  = (float*)         carve((size_t)N * N * 4);   // aliased as A2a later
  unsigned short* An1  = (unsigned short*)carve((size_t)N * N * 2);
  unsigned short* Ah1  = (unsigned short*)carve((size_t)N * N * 2);
  unsigned short* A2an = (unsigned short*)carve((size_t)N * N * 2);
  float*          A2   = (float*)         carve((size_t)P1 * P1 * 4);
  unsigned short* Ah2  = (unsigned short*)carve((size_t)P1 * P1 * 2);
  float*          A3a  = (float*)         carve((size_t)P1 * P1 * 4);
  unsigned short* A3an = (unsigned short*)carve((size_t)P1 * P1 * 2);
  unsigned short* B2n  = (unsigned short*)carve((size_t)P1 * P1 * 2);
  float*          A3   = (float*)         carve((size_t)P2 * P2 * 4);
  unsigned short* Ah3  = (unsigned short*)carve((size_t)P2 * P2 * 2);
  float*          A4a  = (float*)         carve((size_t)P2 * P2 * 4);
  unsigned short* A4an = (unsigned short*)carve((size_t)P2 * P2 * 2);
  unsigned short* B3n  = (unsigned short*)carve((size_t)P2 * P2 * 2);
  float*          dinv = (float*)         carve((size_t)N * 4);
  unsigned short* Wbuf = (unsigned short*)carve((size_t)D * D * 2);
  unsigned short* xbf  = (unsigned short*)carve((size_t)N * D * 2);
  float*          zf   = (float*)         carve((size_t)N * D * 4);
  unsigned short* zbf  = (unsigned short*)carve((size_t)N * D * 2);
  float*          x1   = (float*)         carve((size_t)N * D * 4);
  float*          hbuf = (float*)         carve((size_t)N * D * 4);
  float*          ubuf = (float*)         carve((size_t)N * D * 4);
  float*          x2   = (float*)         carve((size_t)P1 * D * 4);
  float*          x3   = (float*)         carve((size_t)P2 * D * 4);
  float*          x4   = (float*)         carve((size_t)P3 * D * 4);
  float*          scr  = (float*)         carve((size_t)N * 4);
  float*          sval = (float*)         carve((size_t)N * 4);
  float*          pnv  = (float*)         carve(256);
  int*            perm1= (int*)           carve((size_t)P1 * 4);
  int*            perm2= (int*)           carve((size_t)P2 * 4);
  int*            perm3= (int*)           carve((size_t)P3 * 4);
  float*          A2a  = A1f;   // alias: A1f dead once An1/Ah1 built
  if (off > ws_size) return;    // workspace too small; nothing safe to do

  auto nb = [](long n) { return (unsigned)((n + 255) / 256); };
  auto gemm = [&](const unsigned short* A, const unsigned short* B, float* C,
                  const float* bias, int M, int Nn, int K, int zd) {
    dim3 g((unsigned)(M / 128), (unsigned)(Nn / 64));
    k_gemm_bf16<<<g, 256, 0, stream>>>(A, B, C, bias, M, Nn, K, zd);
  };
  // GGAT with concat=True: out[n, 512] (out must differ from xin)
  auto ggat = [&](const unsigned short* An, int n, const float* xin, int Fin,
                  const float* W, const float* b, const float* att, float* out) {
    long xn = (long)n * Fin;
    k_to_bf16<<<nb(xn), 256, 0, stream>>>(xin, xbf, xn);
    long wn = (long)HEADS * Fin * HID;
    k_pack_w<<<nb(wn), 256, 0, stream>>>(W, Wbuf, HEADS, Fin, HID);
    gemm(xbf, Wbuf, zf, b, n, D, Fin, 0);           // z = x@Wcat + b
    long zn = (long)n * D;
    k_to_bf16<<<nb(zn), 256, 0, stream>>>(zf, zbf, zn);
    gemm(An, zbf, out, nullptr, n, D, n, 0);        // out = An @ z
    k_gate_concat<<<(unsigned)(n * HEADS), 128, 0, stream>>>(out, att, n, HEADS, HID);
  };
  auto pool = [&](const float* Aaug, int n, const float* h, const float* p,
                  int k, int* perm, float* xp, float* Ap) {
    k_pnorm<<<1, 256, 0, stream>>>(p, pnv, D);
    k_score<<<(unsigned)n, 256, 0, stream>>>(h, p, pnv, scr, n, D);
    k_topk<<<1, 1024, 0, stream>>>(scr, n, k, perm, sval);
    k_gather_x<<<(unsigned)k, 256, 0, stream>>>(h, perm, sval, xp, k, D);
    if (Ap) k_gather_A<<<(unsigned)k, 256, 0, stream>>>(Aaug, perm, Ap, k, n);
    k_elu<<<nb((long)k * D), 256, 0, stream>>>(xp, (long)k * D);
  };

  // ---- build A1, normalized A1 (== binarize(A1) normalized), Ah1 ----
  k_zero_f32<<<nb((long)N * N), 256, 0, stream>>>(A1f, (long)N * N);
  k_scatter_edges<<<nb(E), 256, 0, stream>>>(A1f, ei, E, N);
  k_deg_rsqrt<<<N, 256, 0, stream>>>(A1f, dinv, N, 0);
  k_norm_bf16<<<nb((long)N * N), 256, 0, stream>>>(A1f, dinv, An1, N, 0);
  k_addI_bf16<<<nb((long)N * N), 256, 0, stream>>>(A1f, Ah1, N);

  // ---- x1 = elu(ggat_c(A1, x)) ----
  ggat(An1, N, x, F_IN, Wc, bc, ac, x1);
  k_elu<<<nb((long)N * D), 256, 0, stream>>>(x1, (long)N * D);

  // ---- down 1: A2a = (A1+I)^2 * (1-I); h = ggat + x1; pool -> x2, A2 ----
  gemm(Ah1, Ah1, A2a, nullptr, N, N, N, 1);         // overwrites A1f (ok)
  k_deg_rsqrt<<<N, 256, 0, stream>>>(A2a, dinv, N, 0);
  k_norm_bf16<<<nb((long)N * N), 256, 0, stream>>>(A2a, dinv, A2an, N, 0);
  ggat(A2an, N, x1, D, Wd1, bd1, ad1, hbuf);
  k_add<<<nb((long)N * D), 256, 0, stream>>>(hbuf, x1, (long)N * D);
  pool(A2a, N, hbuf, pd1, P1, perm1, x2, A2);

  // ---- down 2 ----
  k_addI_bf16<<<nb((long)P1 * P1), 256, 0, stream>>>(A2, Ah2, P1);
  gemm(Ah2, Ah2, A3a, nullptr, P1, P1, P1, 1);
  k_deg_rsqrt<<<P1, 256, 0, stream>>>(A3a, dinv, P1, 0);
  k_norm_bf16<<<nb((long)P1 * P1), 256, 0, stream>>>(A3a, dinv, A3an, P1, 0);
  ggat(A3an, P1, x2, D, Wd2, bd2, ad2, hbuf);
  k_add<<<nb((long)P1 * D), 256, 0, stream>>>(hbuf, x2, (long)P1 * D);
  pool(A3a, P1, hbuf, pd2, P2, perm2, x3, A3);

  // ---- down 3 ----
  k_addI_bf16<<<nb((long)P2 * P2), 256, 0, stream>>>(A3, Ah3, P2);
  gemm(Ah3, Ah3, A4a, nullptr, P2, P2, P2, 1);
  k_deg_rsqrt<<<P2, 256, 0, stream>>>(A4a, dinv, P2, 0);
  k_norm_bf16<<<nb((long)P2 * P2), 256, 0, stream>>>(A4a, dinv, A4an, P2, 0);
  ggat(A4an, P2, x3, D, Wd3, bd3, ad3, hbuf);
  k_add<<<nb((long)P2 * D), 256, 0, stream>>>(hbuf, x3, (long)P2 * D);
  pool(nullptr, P2, hbuf, pd3, P3, perm3, x4, nullptr);   // A4 unused

  // ---- up 1: x3u = elu(ggat(binarize(A3), up3) + x3) ----
  k_zero_f32<<<nb((long)P2 * D), 256, 0, stream>>>(ubuf, (long)P2 * D);
  k_scatter_rows<<<(unsigned)P3, 256, 0, stream>>>(ubuf, perm3, x4, P3, D);
  k_deg_rsqrt<<<P2, 256, 0, stream>>>(A3, dinv, P2, 1);
  k_norm_bf16<<<nb((long)P2 * P2), 256, 0, stream>>>(A3, dinv, B3n, P2, 1);
  ggat(B3n, P2, ubuf, D, Wu1, bu1, au1, hbuf);
  k_add<<<nb((long)P2 * D), 256, 0, stream>>>(hbuf, x3, (long)P2 * D);
  k_elu<<<nb((long)P2 * D), 256, 0, stream>>>(hbuf, (long)P2 * D);   // hbuf = x3u

  // ---- up 2: x2u = elu(ggat(binarize(A2), up2) + x2) ----
  k_zero_f32<<<nb((long)P1 * D), 256, 0, stream>>>(ubuf, (long)P1 * D);
  k_scatter_rows<<<(unsigned)P2, 256, 0, stream>>>(ubuf, perm2, hbuf, P2, D);
  k_deg_rsqrt<<<P1, 256, 0, stream>>>(A2, dinv, P1, 1);
  k_norm_bf16<<<nb((long)P1 * P1), 256, 0, stream>>>(A2, dinv, B2n, P1, 1);
  ggat(B2n, P1, ubuf, D, Wu2, bu2, au2, hbuf);
  k_add<<<nb((long)P1 * D), 256, 0, stream>>>(hbuf, x2, (long)P1 * D);
  k_elu<<<nb((long)P1 * D), 256, 0, stream>>>(hbuf, (long)P1 * D);   // hbuf = x2u

  // ---- up 3 (final, concat=False): out = mean-head ggat(binarize(A1), up1)
  // binarize(A1)==A1 (0/1 entries) -> reuse An1.
  k_zero_f32<<<nb((long)N * D), 256, 0, stream>>>(ubuf, (long)N * D);
  k_scatter_rows<<<(unsigned)P1, 256, 0, stream>>>(ubuf, perm1, hbuf, P1, D);
  k_to_bf16<<<nb((long)N * D), 256, 0, stream>>>(ubuf, xbf, (long)N * D);
  {
    long wn = (long)HEADS * D * 16;
    k_pack_w<<<nb(wn), 256, 0, stream>>>(Wu3, Wbuf, HEADS, D, 16);
  }
  gemm(xbf, Wbuf, zf, bu3, N, HEADS * 16, D, 0);          // z = up1@Wcat + b  [N,64]
  k_to_bf16<<<nb((long)N * 64), 256, 0, stream>>>(zf, zbf, (long)N * 64);
  gemm(An1, zbf, hbuf, nullptr, N, HEADS * 16, N, 0);     // An1 @ z  [N,64]
  k_gate_mean<<<nb(N), 256, 0, stream>>>(hbuf, au3, (float*)d_out, N, HEADS, 16);
}
